// MoEBottleneck_58926951301265
// MI455X (gfx1250) — matrix operand order repaired and checked by reading, not compiled, not run on hardware
//
#include <hip/hip_runtime.h>
#include <hip/hip_bf16.h>

// ---------------------------------------------------------------------------
// MoE bottleneck for MI455X (gfx1250, wave32, WMMA bf16 16x16x32, f32 accum)
// ---------------------------------------------------------------------------

typedef __attribute__((ext_vector_type(16))) __bf16 v16bf;
typedef __attribute__((ext_vector_type(8)))  __bf16 v8bf;
typedef __attribute__((ext_vector_type(8)))  float  v8f;

#define BATCH 64
#define CIN   256
#define WID   64
#define OUT   256
#define HH    56
#define WW    56
#define HW    3136
#define NEXP  4
#define K2    576   // 64 * 9

// padded LDS row strides (multiple of 8 elements => 16B-aligned rows,
// non-power-of-two dword stride => conflict-free half-wave access)
#define LDSA1 264   // conv1 A rows (K=256)
#define LDSB1 264   // conv1 B rows (K=256), transposed [n][k]
#define LDSA2 584   // conv2 A rows (K'=576)
#define LDSIN2 72   // conv2 transposed halo tile: ci stride
#define LDSA3 72    // conv3 A rows (K=64)
#define LDSB3 72    // conv3 B rows (K=64), transposed [n][k]

__device__ __forceinline__ __bf16 f2bf(float f) {
    union { float f; unsigned u; } v; v.f = f;
    unsigned r = (v.u + 0x7FFFu + ((v.u >> 16) & 1u)) >> 16;
    unsigned short s = (unsigned short)r;
    return __builtin_bit_cast(__bf16, s);
}

__device__ __forceinline__ unsigned pack2(float a, float b) {
    unsigned lo = (unsigned)__builtin_bit_cast(unsigned short, f2bf(a));
    unsigned hi = (unsigned)__builtin_bit_cast(unsigned short, f2bf(b));
    return lo | (hi << 16);
}

// A fragment (16x32 bf16): lane-half selects K 0..15 / 16..31; element j maps to
// K = half*16+j  ->  one contiguous 16-elem run.
// p points at &row[k0 + half*16]; runs are at +0 and +8 elements.
__device__ __forceinline__ v16bf frag_a(const __bf16* p) {
    const v8bf* q = (const v8bf*)p;
    v8bf lo = q[0], hi = q[1];
    return __builtin_shufflevector(lo, hi, 0,1,2,3,4,5,6,7,8,9,10,11,12,13,14,15);
}

// B fragment (32x16 bf16): lane-half selects K 0..15 / 16..31; with a
// K-contiguous layout a lane's 16 values are contiguous.
__device__ __forceinline__ v16bf frag_b(const __bf16* p) {
    const v8bf* q = (const v8bf*)p;
    return __builtin_shufflevector(q[0], q[1], 0,1,2,3,4,5,6,7,8,9,10,11,12,13,14,15);
}

// ---------------------------------------------------------------------------
// Row-mean over HW: in[rows][hw] -> out[rows]
// ---------------------------------------------------------------------------
__global__ __launch_bounds__(256) void pool_mean_kernel(
    const float* __restrict__ in, float* __restrict__ out, int hw) {
    __shared__ float s[256];
    const float* p = in + (size_t)blockIdx.x * hw;
    float acc = 0.f;
    for (int i = threadIdx.x; i < hw; i += 256) acc += p[i];
    s[threadIdx.x] = acc;
    __syncthreads();
    for (int st = 128; st > 0; st >>= 1) {
        if (threadIdx.x < st) s[threadIdx.x] += s[threadIdx.x + st];
        __syncthreads();
    }
    if (threadIdx.x == 0) out[blockIdx.x] = s[0] / (float)hw;
}

// ---------------------------------------------------------------------------
// Routing: r[b,e] = mean_{64}( sigmoid(pooled[b,:] @ W.T + bias) )
// ---------------------------------------------------------------------------
__global__ __launch_bounds__(256) void routing_kernel(
    const float* __restrict__ pooled, const float* __restrict__ W,
    const float* __restrict__ bias, float* __restrict__ r, int C) {
    __shared__ float s[256];
    int b = blockIdx.x, j = threadIdx.x;
    const float* pv = pooled + (size_t)b * C;
    const float* wr = W + (size_t)j * C;
    float acc = bias[j];
    for (int i = 0; i < C; ++i) acc += pv[i] * wr[i];
    s[j] = 1.f / (1.f + expf(-acc));
    __syncthreads();
    if (j < NEXP) {
        float a = 0.f;
        for (int t = 0; t < 64; ++t) a += s[j * 64 + t];
        r[b * NEXP + j] = a * (1.f / 64.f);
    }
}

// ---------------------------------------------------------------------------
// Per-sample combined expert weights: w[b,t] = sum_e r[b,e]*ew[e,t] (-> bf16)
// ---------------------------------------------------------------------------
__global__ __launch_bounds__(256) void make_w_kernel(
    const float* __restrict__ r, const float* __restrict__ ew,
    __bf16* __restrict__ w, int per, int total) {
    int idx = blockIdx.x * 256 + threadIdx.x;
    if (idx >= total) return;
    int b = idx / per, t = idx % per;
    float acc = 0.f;
    #pragma unroll
    for (int e = 0; e < NEXP; ++e)
        acc += r[b * NEXP + e] * ew[(size_t)e * per + t];
    w[idx] = f2bf(acc);
}

// ---------------------------------------------------------------------------
// conv2 weights with permuted K:  k' = dy*192 + dx*64 + ci
// (reduction-order permutation applied identically to A and B; GEMM unchanged)
// ---------------------------------------------------------------------------
__global__ __launch_bounds__(256) void make_w2_kernel(
    const float* __restrict__ r, const float* __restrict__ ew,
    __bf16* __restrict__ w) {
    const int per = WID * K2;
    int idx = blockIdx.x * 256 + threadIdx.x;   // grid sized exactly
    int b = idx / per, t = idx % per;
    int o = t / K2, k = t % K2;
    int dy = k / 192, rem = k % 192;
    int dx = rem >> 6, ci = rem & 63;
    int src = o * K2 + ci * 9 + dy * 3 + dx;    // timm layout: o,ci,dy,dx
    float acc = 0.f;
    #pragma unroll
    for (int e = 0; e < NEXP; ++e)
        acc += r[b * NEXP + e] * ew[(size_t)e * per + src];
    w[idx] = f2bf(acc);
}

// ---------------------------------------------------------------------------
// BN folding: sc = g*rsqrt(v+eps), bi = b - m*sc
// ---------------------------------------------------------------------------
__global__ void bnprep_kernel(const float* __restrict__ g, const float* __restrict__ b,
                              const float* __restrict__ m, const float* __restrict__ v,
                              float* __restrict__ sc, float* __restrict__ bi, int n) {
    int i = blockIdx.x * 64 + threadIdx.x;
    if (i < n) {
        float s = g[i] * rsqrtf(v[i] + 1e-5f);
        sc[i] = s;
        bi[i] = b[i] - m[i] * s;
    }
}

// ---------------------------------------------------------------------------
// conv1: per-sample 1x1 GEMM  out1[b,o,n] = sum_i w1[b,o,i] * x[b,i,n]
// M=64, K=256, N tile=64.  grid (B, 49).  + BN1 + ReLU
// ---------------------------------------------------------------------------
__global__ __launch_bounds__(256) void conv1_kernel(
    const float* __restrict__ x, const __bf16* __restrict__ w1,
    const float* __restrict__ sc1, const float* __restrict__ bi1,
    float* __restrict__ out1) {
    __shared__ __bf16 lds_a[WID * LDSA1];   // 33.0 KB, [o][k]
    __shared__ __bf16 lds_b[64 * LDSB1];    // 33.0 KB, [n][k] (transposed)
    int b = blockIdx.x;
    int n0 = blockIdx.y * 64;
    int tid = threadIdx.x;

    const __bf16* wA = w1 + (size_t)b * WID * CIN;
    for (int i = tid; i < WID * 32; i += 256) {
        int o = i >> 5, c = i & 31;
        uint4 val = *(const uint4*)(wA + o * CIN + c * 8);
        *(uint4*)&lds_a[o * LDSA1 + c * 8] = val;
    }
    const float* xb = x + (size_t)b * CIN * HW;
    for (int i = tid; i < (CIN / 2) * 16; i += 256) {
        int kp = i >> 4, n4 = (i & 15) * 4;
        int k = kp * 2;
        const float* p0 = xb + (size_t)k * HW + n0 + n4;
        float4 a0 = *(const float4*)p0;
        float4 a1 = *(const float4*)(p0 + HW);
        *(unsigned*)&lds_b[(n4 + 0) * LDSB1 + k] = pack2(a0.x, a1.x);
        *(unsigned*)&lds_b[(n4 + 1) * LDSB1 + k] = pack2(a0.y, a1.y);
        *(unsigned*)&lds_b[(n4 + 2) * LDSB1 + k] = pack2(a0.z, a1.z);
        *(unsigned*)&lds_b[(n4 + 3) * LDSB1 + k] = pack2(a0.w, a1.w);
    }
    __syncthreads();

    int wave = tid >> 5, lane = tid & 31;
    int mt = wave & 3, nt = wave >> 2;      // 4 M tiles x 2 N groups
    int half = lane >> 4, lcol = lane & 15;
    v8f acc0 = {}, acc1 = {};
    int mrow = mt * 16 + lcol;

    #pragma unroll
    for (int kc = 0; kc < 8; ++kc) {
        int k0 = kc * 32;
        v16bf a = frag_a(&lds_a[mrow * LDSA1 + k0 + half * 16]);
        v16bf b0 = frag_b(&lds_b[(nt * 32 + 0 + lcol) * LDSB1 + k0 + half * 16]);
        v16bf b1 = frag_b(&lds_b[(nt * 32 + 16 + lcol) * LDSB1 + k0 + half * 16]);
        acc0 = __builtin_amdgcn_wmma_f32_16x16x32_bf16(false, a, false, b0, (short)0, acc0, false, false);
        acc1 = __builtin_amdgcn_wmma_f32_16x16x32_bf16(false, a, false, b1, (short)0, acc1, false, false);
    }
    #pragma unroll
    for (int s = 0; s < 2; ++s) {
        v8f acc = s ? acc1 : acc0;
        int n = n0 + nt * 32 + s * 16 + lcol;
        #pragma unroll
        for (int rr = 0; rr < 8; ++rr) {
            int o = mt * 16 + half * 8 + rr;
            float val = fmaf(acc[rr], sc1[o], bi1[o]);
            out1[((size_t)b * WID + o) * HW + n] = val > 0.f ? val : 0.f;
        }
    }
}

// ---------------------------------------------------------------------------
// conv2: per-sample 3x3 implicit GEMM with permuted K' = dy*192 + dx*64 + ci.
// One output row per WG, half of M (32 ch) per WG.  grid (B, 56, 2). +BN2+ReLU
// Halo tile stored transposed: lds_t[(dy*58 + xx)][ci]  (ci contiguous), so a
// lane's 16 B values (16 consecutive ci at fixed dy,dx) are one contiguous run.
// ---------------------------------------------------------------------------
__global__ __launch_bounds__(256) void conv2_kernel(
    const float* __restrict__ in, const __bf16* __restrict__ w2p,
    const float* __restrict__ sc2, const float* __restrict__ bi2,
    float* __restrict__ out2) {
    __shared__ __bf16 lds_a[32 * LDSA2];        // 36.5 KB, [o][k']
    __shared__ __bf16 lds_t[182 * LDSIN2];      // 25.6 KB (8 pad rows for dead lanes)
    int b = blockIdx.x, y = blockIdx.y, mz = blockIdx.z;
    int tid = threadIdx.x;

    const __bf16* wA = w2p + (size_t)b * WID * K2 + (size_t)mz * 32 * K2;
    for (int i = tid; i < 32 * 72; i += 256) {
        int o = i / 72, c = i % 72;
        uint4 val = *(const uint4*)(wA + o * K2 + c * 8);
        *(uint4*)&lds_a[o * LDSA2 + c * 8] = val;
    }
    // transposed halo: lds_t[(dy*58+xx)*72 + ci] = in[ci][y+dy-1][xx-1], ci pairs
    const float* inb = in + (size_t)b * WID * HW;
    for (int i = tid; i < 32 * 192; i += 256) {
        int cp = i / 192, rem = i % 192;
        int dy = rem >> 6, xx = rem & 63;
        if (xx < 58) {
            int ci = cp * 2;
            int gy = y + dy - 1, gx = xx - 1;
            float v0 = 0.f, v1 = 0.f;
            if (gy >= 0 && gy < HH && gx >= 0 && gx < WW) {
                const float* p = inb + (size_t)ci * HW + gy * WW + gx;
                v0 = p[0];
                v1 = p[HW];
            }
            *(unsigned*)&lds_t[(dy * 58 + xx) * LDSIN2 + ci] = pack2(v0, v1);
        }
    }
    __syncthreads();

    int wave = tid >> 5, lane = tid & 31;
    int mt = wave & 1, nt = wave >> 1;      // 2 M tiles x 4 N tiles
    int half = lane >> 4, lcol = lane & 15;
    int x = nt * 16 + lcol;
    int mrow = mt * 16 + lcol;
    v8f acc = {};

    #pragma unroll
    for (int kc = 0; kc < 18; ++kc) {
        int k0 = kc * 32;
        v16bf a = frag_a(&lds_a[mrow * LDSA2 + k0 + half * 16]);
        int kh = k0 + half * 16;            // this half's K' run start
        int blk = kh >> 6;                  // (dy,dx) block index
        int ci0 = kh & 63;
        int dy = blk / 3, dx = blk - dy * 3;
        v16bf bf = frag_b(&lds_t[(dy * 58 + x + dx) * LDSIN2 + ci0]);
        acc = __builtin_amdgcn_wmma_f32_16x16x32_bf16(false, a, false, bf, (short)0, acc, false, false);
    }
    if (x < WW) {
        #pragma unroll
        for (int rr = 0; rr < 8; ++rr) {
            int o = mz * 32 + mt * 16 + half * 8 + rr;
            float val = fmaf(acc[rr], sc2[o], bi2[o]);
            out2[((size_t)b * WID + o) * HW + y * WW + x] = val > 0.f ? val : 0.f;
        }
    }
}

// ---------------------------------------------------------------------------
// conv3: shared-weight 1x1 GEMM  out[b,o,n] = sum_i w3[o,i]*out2[b,i,n]
// M=256, K=64, N tile=64.  grid (B, 49).  + BN3 + residual + ReLU -> d_out
// ---------------------------------------------------------------------------
__global__ __launch_bounds__(256) void conv3_kernel(
    const float* __restrict__ in, const float* __restrict__ w3,
    const float* __restrict__ sc3, const float* __restrict__ bi3,
    const float* __restrict__ identity, float* __restrict__ out) {
    __shared__ __bf16 lds_a[OUT * LDSA3];   // 36 KB, [o][k]
    __shared__ __bf16 lds_b[64 * LDSB3];    // 9 KB,  [n][k] (transposed)
    int b = blockIdx.x;
    int n0 = blockIdx.y * 64;
    int tid = threadIdx.x;

    for (int i = tid; i < OUT * 16; i += 256) {
        int o = i >> 4, c = i & 15;
        float4 f = *(const float4*)(w3 + o * WID + c * 4);
        unsigned* dst = (unsigned*)&lds_a[o * LDSA3 + c * 4];
        dst[0] = pack2(f.x, f.y);
        dst[1] = pack2(f.z, f.w);
    }
    const float* inb = in + (size_t)b * WID * HW;
    for (int i = tid; i < (WID / 2) * 16; i += 256) {
        int kp = i >> 4, n4 = (i & 15) * 4;
        int k = kp * 2;
        const float* p0 = inb + (size_t)k * HW + n0 + n4;
        float4 a0 = *(const float4*)p0;
        float4 a1 = *(const float4*)(p0 + HW);
        *(unsigned*)&lds_b[(n4 + 0) * LDSB3 + k] = pack2(a0.x, a1.x);
        *(unsigned*)&lds_b[(n4 + 1) * LDSB3 + k] = pack2(a0.y, a1.y);
        *(unsigned*)&lds_b[(n4 + 2) * LDSB3 + k] = pack2(a0.z, a1.z);
        *(unsigned*)&lds_b[(n4 + 3) * LDSB3 + k] = pack2(a0.w, a1.w);
    }
    __syncthreads();

    int wave = tid >> 5, lane = tid & 31;
    int half = lane >> 4, lcol = lane & 15;
    v8f acc[8] = {};  // 2 M tiles (wave, wave+8) x 4 N tiles

    #pragma unroll
    for (int kc = 0; kc < 2; ++kc) {
        int k0 = kc * 32;
        v16bf bfr[4];
        #pragma unroll
        for (int s = 0; s < 4; ++s)
            bfr[s] = frag_b(&lds_b[(s * 16 + lcol) * LDSB3 + k0 + half * 16]);
        #pragma unroll
        for (int mi = 0; mi < 2; ++mi) {
            int mrow = (wave + mi * 8) * 16 + lcol;
            v16bf a = frag_a(&lds_a[mrow * LDSA3 + k0 + half * 16]);
            #pragma unroll
            for (int s = 0; s < 4; ++s)
                acc[mi * 4 + s] = __builtin_amdgcn_wmma_f32_16x16x32_bf16(
                    false, a, false, bfr[s], (short)0, acc[mi * 4 + s], false, false);
        }
    }
    #pragma unroll
    for (int mi = 0; mi < 2; ++mi) {
        #pragma unroll
        for (int s = 0; s < 4; ++s) {
            int n = n0 + s * 16 + lcol;
            #pragma unroll
            for (int rr = 0; rr < 8; ++rr) {
                int o = (wave + mi * 8) * 16 + half * 8 + rr;
                size_t off = ((size_t)b * OUT + o) * HW + n;
                float val = fmaf(acc[mi * 4 + s][rr], sc3[o], bi3[o]) + identity[off];
                out[off] = val > 0.f ? val : 0.f;
            }
        }
    }
}

// ---------------------------------------------------------------------------
extern "C" void kernel_launch(void* const* d_in, const int* in_sizes, int n_in,
                              void* d_out, int out_size, void* d_ws, size_t ws_size,
                              hipStream_t stream) {
    const float* x     = (const float*)d_in[0];
    const float* r1_W  = (const float*)d_in[1];
    const float* r1_b  = (const float*)d_in[2];
    const float* ew1   = (const float*)d_in[3];
    const float* bn1_g = (const float*)d_in[4];
    const float* bn1_b = (const float*)d_in[5];
    const float* bn1_m = (const float*)d_in[6];
    const float* bn1_v = (const float*)d_in[7];
    const float* r2_W  = (const float*)d_in[8];
    const float* r2_b  = (const float*)d_in[9];
    const float* ew2   = (const float*)d_in[10];
    const float* bn2_g = (const float*)d_in[11];
    const float* bn2_b = (const float*)d_in[12];
    const float* bn2_m = (const float*)d_in[13];
    const float* bn2_v = (const float*)d_in[14];
    const float* w3    = (const float*)d_in[15];
    const float* bn3_g = (const float*)d_in[16];
    const float* bn3_b = (const float*)d_in[17];
    const float* bn3_m = (const float*)d_in[18];
    const float* bn3_v = (const float*)d_in[19];

    float* ws      = (float*)d_ws;
    float* pooled1 = ws;                                     // 64*256
    float* r1      = pooled1 + BATCH * CIN;                  // 64*4
    float* pooled2 = r1 + BATCH * NEXP;                      // 64*64
    float* r2      = pooled2 + BATCH * WID;                  // 64*4
    float* sc1     = r2 + BATCH * NEXP;                      // 64
    float* bi1     = sc1 + WID;                              // 64
    float* sc2     = bi1 + WID;                              // 64
    float* bi2     = sc2 + WID;                              // 64
    float* sc3     = bi2 + WID;                              // 256
    float* bi3     = sc3 + OUT;                              // 256
    float* out1    = bi3 + OUT;                              // 64*64*3136
    float* out2    = out1 + (size_t)BATCH * WID * HW;        // 64*64*3136
    __bf16* w1     = (__bf16*)(out2 + (size_t)BATCH * WID * HW);  // 64*64*256
    __bf16* w2p    = w1 + (size_t)BATCH * WID * CIN;              // 64*64*576

    // fold all three BNs
    bnprep_kernel<<<1, 64, 0, stream>>>(bn1_g, bn1_b, bn1_m, bn1_v, sc1, bi1, WID);
    bnprep_kernel<<<1, 64, 0, stream>>>(bn2_g, bn2_b, bn2_m, bn2_v, sc2, bi2, WID);
    bnprep_kernel<<<4, 64, 0, stream>>>(bn3_g, bn3_b, bn3_m, bn3_v, sc3, bi3, OUT);

    // stage 1: routed 1x1 CondConv (256->64) + BN + ReLU
    pool_mean_kernel<<<BATCH * CIN, 256, 0, stream>>>(x, pooled1, HW);
    routing_kernel<<<BATCH, 256, 0, stream>>>(pooled1, r1_W, r1_b, r1, CIN);
    {
        int per = WID * CIN, total = BATCH * per;
        make_w_kernel<<<(total + 255) / 256, 256, 0, stream>>>(r1, ew1, w1, per, total);
    }
    conv1_kernel<<<dim3(BATCH, HW / 64), 256, 0, stream>>>(x, w1, sc1, bi1, out1);

    // stage 2: routed 3x3 CondConv (64->64) + BN + ReLU
    pool_mean_kernel<<<BATCH * WID, 256, 0, stream>>>(out1, pooled2, HW);
    routing_kernel<<<BATCH, 256, 0, stream>>>(pooled2, r2_W, r2_b, r2, WID);
    make_w2_kernel<<<(BATCH * WID * K2) / 256, 256, 0, stream>>>(r2, ew2, w2p);
    conv2_kernel<<<dim3(BATCH, HH, 2), 256, 0, stream>>>(out1, w2p, sc2, bi2, out2);

    // stage 3: shared 1x1 (64->256) + BN + residual + ReLU
    conv3_kernel<<<dim3(BATCH, HW / 64), 256, 0, stream>>>(
        out2, w3, sc3, bi3, x, (float*)d_out);
}
